// EMAMultiheadAttention_19722489823288
// MI455X (gfx1250) — matrix operand order repaired and checked by reading, not compiled, not run on hardware
//
#include <hip/hip_runtime.h>
#include <hip/hip_bf16.h>
#include <math.h>

// ---------------------------------------------------------------------------
// CDNA5 (gfx1250) implementation of the MEGA-style EMA attention block.
// Matmuls: V_WMMA_F32_16X16X32_BF16 fed from LDS tiles staged with
// GLOBAL_LOAD_ASYNC_TO_LDS_B128 (ASYNCcnt) — cooperative 128x128 block tiles.
// ---------------------------------------------------------------------------

typedef __attribute__((ext_vector_type(16))) __bf16 v16bf;
typedef __attribute__((ext_vector_type(8)))  float  v8f;

#define EMA_K 15
#define TILE_PAD 40                 // halves per LDS tile row (80B: conflict-free, 16B aligned)
#define TILE_HALFS (128 * TILE_PAD) // 5120 halves per 128x32 tile
#define SMEM_BYTES (4 * TILE_HALFS * 2) // A0,A1,B0,B1 = 40 KB

__device__ __forceinline__ float sigmoidf_(float v) { return 1.0f / (1.0f + __expf(-v)); }
__device__ __forceinline__ float siluf_(float v)    { return v / (1.0f + __expf(-v)); }

// float -> bf16 (round to nearest even), stored as u16
__device__ __forceinline__ unsigned short f2bf(float f) {
    unsigned u = __float_as_uint(f);
    unsigned r = u + 0x7FFFu + ((u >> 16) & 1u);
    return (unsigned short)(r >> 16);
}

// ---------------------------------------------------------------------------
// CDNA5 async copy: 16 bytes global -> LDS per lane, tracked by ASYNCcnt.
// (inline asm: portable across ROCm7.2 / amdgpu-toolchain builtin arity skew)
// ---------------------------------------------------------------------------
__device__ __forceinline__ void async_b128(unsigned lds_off, unsigned long long gaddr) {
    asm volatile("global_load_async_to_lds_b128 %0, %1, off"
                 :: "v"(lds_off), "v"(gaddr) : "memory");
}
__device__ __forceinline__ void wait_asynccnt0() {
    asm volatile("s_wait_asynccnt 0x0" ::: "memory");
}
__device__ __forceinline__ unsigned lds_offset(const void* p) {
    return (unsigned)(uintptr_t)p;  // generic LDS pointer: addr[31:0] == LDS offset
}

// ---------------------------------------------------------------------------
// Fragment loaders from LDS tiles (ISA 7.12.2 layouts, wave32)
// Tiles are [128 rows][32 halves] row-major, row stride TILE_PAD halves.
// A 16x32: lanes 0-15 row=lane,   K {0..7, 16..23}; lanes16-31 K {8..15, 24..31}
// ---------------------------------------------------------------------------
__device__ __forceinline__ v16bf lds_frag_a(const unsigned short* T, int r0, int lane) {
    const int r    = r0 + (lane & 15);
    const int koff = (lane >> 4) ? 8 : 0;
    const unsigned short* base = T + r * TILE_PAD + koff;
    union { v16bf v; unsigned u[8]; } f;
#pragma unroll
    for (int i = 0; i < 4; ++i)
        f.u[i] = *(const unsigned*)(const void*)(base + 2 * i);
#pragma unroll
    for (int i = 0; i < 4; ++i)
        f.u[4 + i] = *(const unsigned*)(const void*)(base + 16 + 2 * i);
    return f.v;
}

// B 32x16: lanes 0-15 N=lane (K {0..7, 16..23}), lanes16-31 N=lane-16 (K {8..15, 24..31})
__device__ __forceinline__ v16bf lds_frag_b(const unsigned short* T, int n0, int lane) {
    const int n  = n0 + (lane & 15);
    const int kb = (lane >> 4) << 3;
    const unsigned short* base = T + n * TILE_PAD + kb;
    union { v16bf v; uint4 q[2]; } f;
    f.q[0] = *(const uint4*)(const void*)(base);
    f.q[1] = *(const uint4*)(const void*)(base + 16);
    return f.v;
}

// Stage one K-step: A[128 x 32] and B[128 x 32] tiles -> LDS (async).
// 256 threads x 2 chunks x 16B per tile.
__device__ __forceinline__ void stage_tiles(unsigned short* sA, unsigned short* sB,
                                            const unsigned short* Ag, long long lda, int mBase,
                                            const unsigned short* Bg, long long ldb, int nBase,
                                            int k0, int t) {
#pragma unroll
    for (int i = 0; i < 2; ++i) {
        const int c   = t + i * 256;
        const int row = c >> 2;
        const int col = (c & 3) * 8;
        async_b128(lds_offset(sA + row * TILE_PAD + col),
                   (unsigned long long)(uintptr_t)(Ag + (size_t)(mBase + row) * (size_t)lda + k0 + col));
        async_b128(lds_offset(sB + row * TILE_PAD + col),
                   (unsigned long long)(uintptr_t)(Bg + (size_t)(nBase + row) * (size_t)ldb + k0 + col));
    }
}

// ---------------------------------------------------------------------------
// Cooperative WMMA GEMM: C[M,N] = act( scale * A[M,K] * Bsrc[N,K]^T + bias[N] )
//  block = 256 threads (8 waves), block tile 128x128, wave tile 32x64
//  double-buffered async LDS staging; M,N % 128 == 0, K % 32 == 0
// act: 0 = none, 1 = silu, 2 = sigmoid
// ---------------------------------------------------------------------------
__global__ void wmma_gemm_bf16(const unsigned short* __restrict__ A, long long lda,
                               const unsigned short* __restrict__ Bm, long long ldb,
                               const float* __restrict__ bias, float scale, int act,
                               float* __restrict__ outf,
                               unsigned short* __restrict__ outb, int outb_trans, long long ldo,
                               int M, int N, int Kd) {
    extern __shared__ unsigned short smem[];
    // buffers: [0]=A0 [1]=A1 [2]=B0 [3]=B1 (computed per-use; no LDS ptr aggregates)

    const int t    = threadIdx.x;
    const int lane = t & 31;
    const int wv   = t >> 5;             // 0..7
    const int ntc  = N >> 7;             // 128-wide N tiles
    const int mt   = blockIdx.x / ntc;
    const int nt   = blockIdx.x % ntc;
    const int mBase = mt << 7, nBase = nt << 7;
    const int wm = wv >> 1;              // 0..3 -> 32-row strip
    const int wn = wv & 1;               // 0..1 -> 64-col strip

    v8f acc[2][4] = {};                  // 2 m-tiles x 4 n-tiles of 16x16 f32

    const int KT = Kd >> 5;              // K steps of 32
    stage_tiles(smem, smem + 2 * TILE_HALFS, A, lda, mBase, Bm, ldb, nBase, 0, t);

    for (int kt = 0; kt < KT; ++kt) {
        wait_asynccnt0();                // our async copies into buf kt&1 done
        __syncthreads();                 // everyone's copies done; prev compute done
        const int buf = kt & 1;
        unsigned short* curA = smem + buf * TILE_HALFS;
        unsigned short* curB = smem + (2 + buf) * TILE_HALFS;
        if (kt + 1 < KT) {               // prefetch next K step into other buffer
            unsigned short* nxtA = smem + (buf ^ 1) * TILE_HALFS;
            unsigned short* nxtB = smem + (2 + (buf ^ 1)) * TILE_HALFS;
            stage_tiles(nxtA, nxtB, A, lda, mBase, Bm, ldb, nBase, (kt + 1) << 5, t);
        }

        v16bf af0 = lds_frag_a(curA, wm * 32,      lane);
        v16bf af1 = lds_frag_a(curA, wm * 32 + 16, lane);
#pragma unroll
        for (int jn = 0; jn < 4; ++jn) {
            v16bf bf = lds_frag_b(curB, wn * 64 + jn * 16, lane);
            acc[0][jn] = __builtin_amdgcn_wmma_f32_16x16x32_bf16(
                false, af0, false, bf, (short)0, acc[0][jn], false, false);
            acc[1][jn] = __builtin_amdgcn_wmma_f32_16x16x32_bf16(
                false, af1, false, bf, (short)0, acc[1][jn], false, false);
        }
    }

    // Epilogue per ISA C/D layout: VGPR r -> M = +r + 8*(lane>=16), N = +(lane&15)
    const int nn  = lane & 15;
    const int mhi = (lane >> 4) << 3;
#pragma unroll
    for (int jm = 0; jm < 2; ++jm) {
#pragma unroll
        for (int jn = 0; jn < 4; ++jn) {
            const int n  = nBase + wn * 64 + jn * 16 + nn;
            const float bv = bias ? bias[n] : 0.0f;
#pragma unroll
            for (int r = 0; r < 8; ++r) {
                const int m = mBase + wm * 32 + jm * 16 + mhi + r;
                float v = acc[jm][jn][r] * scale + bv;
                if (act == 1)      v = siluf_(v);
                else if (act == 2) v = sigmoidf_(v);
                if (outf) outf[(size_t)m * N + n] = v;
                if (outb) {
                    if (outb_trans) outb[(size_t)n * (size_t)ldo + m] = f2bf(v);
                    else            outb[(size_t)m * (size_t)ldo + n] = f2bf(v);
                }
            }
        }
    }
}

// ---------------------------------------------------------------------------
// EMA (depthwise causal 15-tap conv) + silu(x*omega + ema)  -> x', plus
// masked x in f32 and bf16. One thread per (b, d) channel scans S.
// ---------------------------------------------------------------------------
__global__ void ema_xprime_kernel(const float* __restrict__ q,
                                  const unsigned char* __restrict__ mask,
                                  const float* __restrict__ omega,
                                  const float* __restrict__ delta,
                                  const float* __restrict__ alpha,
                                  const float* __restrict__ beta,
                                  const float* __restrict__ gamma,
                                  float* __restrict__ xf,
                                  unsigned short* __restrict__ xbf,
                                  unsigned short* __restrict__ xpbf,
                                  int S, int D, int Ne) {
    const int d = blockIdx.x * blockDim.x + threadIdx.x;
    const int b = blockIdx.y;
    if (d >= D) return;

    float w[EMA_K];
#pragma unroll
    for (int j = 0; j < EMA_K; ++j) w[j] = 0.0f;
    for (int n = 0; n < Ne; ++n) {
        const float p  = sigmoidf_(delta[d * Ne + n]);
        const float a  = sigmoidf_(alpha[d * Ne + n]);
        const float c  = p * beta[d * Ne + n] * gamma[d * Ne + n];
        const float qq = 1.0f - p * a;
        float pw = 1.0f;
#pragma unroll
        for (int j = 0; j < EMA_K; ++j) { w[j] += c * pw; pw *= qq; }
    }

    float hist[EMA_K];
#pragma unroll
    for (int j = 0; j < EMA_K; ++j) hist[j] = 0.0f;
    const float om = omega[d];

    for (int s = 0; s < S; ++s) {
        const size_t idx = ((size_t)b * S + s) * D + d;
        const float xv = mask[(size_t)b * S + s] ? q[idx] : 0.0f;
#pragma unroll
        for (int j = EMA_K - 1; j > 0; --j) hist[j] = hist[j - 1];
        hist[0] = xv;
        float acc = 0.0f;
#pragma unroll
        for (int j = 0; j < EMA_K; ++j) acc += w[j] * hist[j];
        const float xp = siluf_(acc + xv * om);
        xf[idx]   = xv;
        xbf[idx]  = f2bf(xv);
        xpbf[idx] = f2bf(xp);
    }
}

// ---------------------------------------------------------------------------
// RoPE for Q and K: reads z (f32), writes bf16 Q,K.  one thread per (b,s,h<half)
// ---------------------------------------------------------------------------
__global__ void rope_qk_kernel(const float* __restrict__ z,
                               const float* __restrict__ kapQ, const float* __restrict__ muQ,
                               const float* __restrict__ kapK, const float* __restrict__ muK,
                               unsigned short* __restrict__ Qb, unsigned short* __restrict__ Kb,
                               int S, int D, long long total) {
    const long long idx = (long long)blockIdx.x * blockDim.x + threadIdx.x;
    if (idx >= total) return;
    const int half = D >> 1;
    const int h = (int)(idx % half);
    const long long bs = idx / half;
    const int s = (int)(bs % S);
    const size_t base = (size_t)bs * D;

    const float ang = (float)s * __powf(10000.0f, -(float)h / (float)half);
    const float c = __cosf(ang), sn = __sinf(ang);

    const float z1 = z[base + h], z2 = z[base + h + half];

    const float q1 = kapQ[h] * z1 + muQ[h];
    const float q2 = kapQ[h + half] * z2 + muQ[h + half];
    Qb[base + h]        = f2bf(q1 * c - q2 * sn);
    Qb[base + h + half] = f2bf(q1 * sn + q2 * c);

    const float k1 = kapK[h] * z1 + muK[h];
    const float k2 = kapK[h + half] * z2 + muK[h + half];
    Kb[base + h]        = f2bf(k1 * c - k2 * sn);
    Kb[base + h + half] = f2bf(k1 * sn + k2 * c);
}

// ---------------------------------------------------------------------------
// Row softmax with key mask: scores[b,q,:] (f32) -> P[b,q,:] (bf16)
// ---------------------------------------------------------------------------
__global__ void softmax_kernel(const float* __restrict__ scores,
                               const unsigned char* __restrict__ mask,
                               unsigned short* __restrict__ P, int S) {
    __shared__ float red[256];
    const int tid = threadIdx.x;
    const int row = blockIdx.x, b = blockIdx.y;
    const float* src = scores + ((size_t)b * S + row) * S;
    unsigned short* dst = P + ((size_t)b * S + row) * S;
    const unsigned char* mk = mask + (size_t)b * S;

    float mx = -INFINITY;
    for (int c = tid; c < S; c += 256)
        if (mk[c]) mx = fmaxf(mx, src[c]);
    red[tid] = mx; __syncthreads();
    for (int st = 128; st > 0; st >>= 1) {
        if (tid < st) red[tid] = fmaxf(red[tid], red[tid + st]);
        __syncthreads();
    }
    mx = red[0]; __syncthreads();
    const bool valid = (mx > -INFINITY);

    float sum = 0.0f;
    for (int c = tid; c < S; c += 256)
        if (valid && mk[c]) sum += __expf(src[c] - mx);
    red[tid] = sum; __syncthreads();
    for (int st = 128; st > 0; st >>= 1) {
        if (tid < st) red[tid] += red[tid + st];
        __syncthreads();
    }
    sum = red[0];
    const float inv = (valid && sum > 0.0f) ? 1.0f / sum : 0.0f;

    for (int c = tid; c < S; c += 256) {
        float pv = (valid && mk[c]) ? __expf(src[c] - mx) * inv : 0.0f;
        dst[c] = f2bf(pv);
    }
}

// ---------------------------------------------------------------------------
// Final gate: out = phi*silu(hpre + gamma*ou) + (1-phi)*x
// ---------------------------------------------------------------------------
__global__ void final_gate_kernel(const float* __restrict__ hpre,
                                  const float* __restrict__ gamma,
                                  const float* __restrict__ ou,
                                  const float* __restrict__ phi,
                                  const float* __restrict__ x,
                                  float* __restrict__ out, long long n) {
    const long long i = (long long)blockIdx.x * blockDim.x + threadIdx.x;
    if (i >= n) return;
    const float hat = siluf_(hpre[i] + gamma[i] * ou[i]);
    const float ph  = phi[i];
    out[i] = ph * hat + (1.0f - ph) * x[i];
}

__global__ void f32_to_bf16_kernel(const float* __restrict__ in,
                                   unsigned short* __restrict__ out, long long n) {
    const long long i = (long long)blockIdx.x * blockDim.x + threadIdx.x;
    if (i < n) out[i] = f2bf(in[i]);
}

// ---------------------------------------------------------------------------
// Host launcher
// ---------------------------------------------------------------------------
extern "C" void kernel_launch(void* const* d_in, const int* in_sizes, int n_in,
                              void* d_out, int out_size, void* d_ws, size_t ws_size,
                              hipStream_t stream) {
    (void)in_sizes; (void)n_in; (void)out_size; (void)ws_size;
    const int Bc = 4, Sc = 2048, Dc = 1024, Ne = 16;
    const long long BS  = (long long)Bc * Sc;        // 8192
    const long long BSD = BS * Dc;                   // 8,388,608
    const long long DD  = (long long)Dc * Dc;

    const float*         queries = (const float*)d_in[0];
    const unsigned char* mask    = (const unsigned char*)d_in[1];
    const float* omega   = (const float*)d_in[2];
    const float* Wz      = (const float*)d_in[3];
    const float* bz      = (const float*)d_in[4];
    const float* kapQ    = (const float*)d_in[5];
    const float* muQ     = (const float*)d_in[6];
    const float* kapK    = (const float*)d_in[7];
    const float* muK     = (const float*)d_in[8];
    const float* Wv      = (const float*)d_in[9];
    const float* bv      = (const float*)d_in[10];
    const float* Wg      = (const float*)d_in[11];
    const float* bg      = (const float*)d_in[12];
    const float* Wp      = (const float*)d_in[13];
    const float* bp      = (const float*)d_in[14];
    const float* Wh      = (const float*)d_in[15];
    const float* bh      = (const float*)d_in[16];
    const float* Uh      = (const float*)d_in[17];
    const float* e_delta = (const float*)d_in[18];
    const float* e_alpha = (const float*)d_in[19];
    const float* e_beta  = (const float*)d_in[20];
    const float* e_gamma = (const float*)d_in[21];
    float* out = (float*)d_out;

    // workspace layout (bytes); aliased regions recycled after attention.
    char* ws = (char*)d_ws;
    const size_t MB = 1ull << 20;
    float*          x_f32  = (float*)(ws + 0 * MB);            // 32 MB
    unsigned short* xp_bf  = (unsigned short*)(ws + 32 * MB);  // 16 MB
    unsigned short* x_bf   = (unsigned short*)(ws + 48 * MB);  // 16 MB
    float*          z_f32  = (float*)(ws + 64 * MB);           // 32 MB (reused as phi)
    unsigned short* Q_bf   = (unsigned short*)(ws + 96 * MB);  // 16 MB
    unsigned short* K_bf   = (unsigned short*)(ws + 112 * MB); // 16 MB
    unsigned short* Vt_bf  = (unsigned short*)(ws + 128 * MB); // 16 MB (V^T [D, B*S])
    unsigned short* O_bf   = (unsigned short*)(ws + 144 * MB); // 16 MB
    float*          scores = (float*)(ws + 160 * MB);          // 64 MB (reused: gamma@160, OU@192)
    unsigned short* P_bf   = (unsigned short*)(ws + 224 * MB); // 32 MB (reused as hpre)
    float*          gammaB = (float*)(ws + 160 * MB);
    float*          ouB    = (float*)(ws + 192 * MB);
    float*          hpreB  = (float*)(ws + 224 * MB);
    float*          phiB   = z_f32;
    unsigned short* Wz_bf  = (unsigned short*)(ws + 256 * MB); // 2 MB each
    unsigned short* Wv_bf  = (unsigned short*)(ws + 258 * MB);
    unsigned short* Wg_bf  = (unsigned short*)(ws + 260 * MB);
    unsigned short* Wp_bf  = (unsigned short*)(ws + 262 * MB);
    unsigned short* Wh_bf  = (unsigned short*)(ws + 264 * MB);
    unsigned short* Uh_bf  = (unsigned short*)(ws + 266 * MB); // total ~268 MB

    // --- 1. weights -> bf16 ---
    {
        const int blk = 256;
        const int grd = (int)((DD + blk - 1) / blk);
        f32_to_bf16_kernel<<<grd, blk, 0, stream>>>(Wz, Wz_bf, DD);
        f32_to_bf16_kernel<<<grd, blk, 0, stream>>>(Wv, Wv_bf, DD);
        f32_to_bf16_kernel<<<grd, blk, 0, stream>>>(Wg, Wg_bf, DD);
        f32_to_bf16_kernel<<<grd, blk, 0, stream>>>(Wp, Wp_bf, DD);
        f32_to_bf16_kernel<<<grd, blk, 0, stream>>>(Wh, Wh_bf, DD);
        f32_to_bf16_kernel<<<grd, blk, 0, stream>>>(Uh, Uh_bf, DD);
    }

    // --- 2. EMA + x' ---
    ema_xprime_kernel<<<dim3(Dc / 256, Bc), 256, 0, stream>>>(
        queries, mask, omega, e_delta, e_alpha, e_beta, e_gamma,
        x_f32, x_bf, xp_bf, Sc, Dc, Ne);

    auto gemm = [&](const unsigned short* A, long long lda,
                    const unsigned short* Bm, long long ldb,
                    const float* bias, float scale, int act,
                    float* of, unsigned short* ob, int obT, long long ldo,
                    int M, int N, int Kd) {
        const int blocks = (M >> 7) * (N >> 7);
        wmma_gemm_bf16<<<blocks, 256, SMEM_BYTES, stream>>>(
            A, lda, Bm, ldb, bias, scale, act, of, ob, obT, ldo, M, N, Kd);
    };

    // --- 3. z = silu(x' Wz^T + bz) ---
    gemm(xp_bf, Dc, Wz_bf, Dc, bz, 1.0f, 1, z_f32, nullptr, 0, 0, (int)BS, Dc, Dc);

    // --- 4. RoPE -> Q, K (bf16) ---
    {
        const long long total = BS * (Dc / 2);
        rope_qk_kernel<<<(int)((total + 255) / 256), 256, 0, stream>>>(
            z_f32, kapQ, muQ, kapK, muK, Q_bf, K_bf, Sc, Dc, total);
    }

    // --- 5. V = silu(x Wv^T + bv), stored transposed [D, B*S] for PV ---
    gemm(x_bf, Dc, Wv_bf, Dc, bv, 1.0f, 1, nullptr, Vt_bf, 1, BS, (int)BS, Dc, Dc);

    // --- 6. attention per batch ---
    const float att_scale = 1.0f / sqrtf((float)Dc);
    for (int b = 0; b < Bc; ++b) {
        gemm(Q_bf + (size_t)b * Sc * Dc, Dc,
             K_bf + (size_t)b * Sc * Dc, Dc,
             nullptr, att_scale, 0,
             scores + (size_t)b * Sc * Sc, nullptr, 0, 0, Sc, Sc, Dc);
    }
    softmax_kernel<<<dim3(Sc, Bc), 256, 0, stream>>>(scores, mask, P_bf, Sc);
    for (int b = 0; b < Bc; ++b) {
        gemm(P_bf + (size_t)b * Sc * Sc, Sc,
             Vt_bf + (size_t)b * Sc, BS,
             nullptr, 1.0f, 0,
             nullptr, O_bf + (size_t)b * Sc * Dc, 0, Dc, Sc, Dc, Sc);
    }

    // --- 7. post-attention linears (scores/P regions recycled) ---
    gemm(O_bf,  Dc, Uh_bf, Dc, nullptr, 1.0f, 0, ouB,    nullptr, 0, 0, (int)BS, Dc, Dc);
    gemm(xp_bf, Dc, Wg_bf, Dc, bg,      1.0f, 1, gammaB, nullptr, 0, 0, (int)BS, Dc, Dc);
    gemm(xp_bf, Dc, Wh_bf, Dc, bh,      1.0f, 0, hpreB,  nullptr, 0, 0, (int)BS, Dc, Dc);
    gemm(xp_bf, Dc, Wp_bf, Dc, bp,      1.0f, 2, phiB,   nullptr, 0, 0, (int)BS, Dc, Dc);

    // --- 8. out = phi*silu(hpre + gamma*ou) + (1-phi)*x ---
    final_gate_kernel<<<(int)((BSD + 255) / 256), 256, 0, stream>>>(
        hpreB, gammaB, ouB, phiB, x_f32, out, BSD);
}
